// SpatialAttention_41807211659721
// MI455X (gfx1250) — matrix operand order repaired and checked by reading, not compiled, not run on hardware
//
#include <hip/hip_runtime.h>
#include <hip/hip_bf16.h>

// ---- CDNA5 (gfx1250) vector types ------------------------------------------
typedef __attribute__((ext_vector_type(16))) _Float16 v16h;
typedef __attribute__((ext_vector_type(8)))  float    v8f;
typedef __attribute__((ext_vector_type(4)))  float    f4;

#define B_   16
#define C_   256
#define H_   128
#define W_   128
#define HW_  (H_ * W_)          // 16384
#define NPIX (B_ * HW_)         // 262144

// ----------------------------------------------------------------------------
// Kernel 1: channel-wise mean + max reduction over C=256.
// 65536 threads, each owns 4 consecutive pixels (float4 / b128 loads).
// Lanes are consecutive pixels -> fully coalesced.
// Non-temporal loads: x (268MB) is streamed once, keep L2 for feat.
// ----------------------------------------------------------------------------
__global__ __launch_bounds__(256) void
sa_reduce_mean_max(const float* __restrict__ x, float* __restrict__ feat) {
    const int t  = blockIdx.x * 256 + threadIdx.x;   // 0 .. 65535
    const int b  = t >> 12;                          // 4096 threads per image
    const int p4 = (t & 4095) << 2;                  // float offset within image

    const f4* __restrict__ xb =
        (const f4*)(x + (size_t)b * C_ * HW_ + p4);  // channel stride = HW_/4 f4

    f4 s; s.x = 0.f; s.y = 0.f; s.z = 0.f; s.w = 0.f;
    f4 m; m.x = m.y = m.z = m.w = -3.402823466e+38f;

    #pragma unroll 16
    for (int c = 0; c < C_; ++c) {
        // prefetch ~2 MB of stream ahead, uniform branch
        if ((c & 15) == 0 && (c + 32) < C_)
            __builtin_prefetch(&xb[(size_t)(c + 32) * (HW_ / 4)], 0, 0);
        f4 v = __builtin_nontemporal_load(&xb[(size_t)c * (HW_ / 4)]);
        s.x += v.x; s.y += v.y; s.z += v.z; s.w += v.w;
        m.x = fmaxf(m.x, v.x); m.y = fmaxf(m.y, v.y);
        m.z = fmaxf(m.z, v.z); m.w = fmaxf(m.w, v.w);
    }

    const float inv = 1.0f / (float)C_;
    f4 a; a.x = s.x * inv; a.y = s.y * inv; a.z = s.z * inv; a.w = s.w * inv;

    float* fb = feat + (size_t)b * 2 * HW_ + p4;     // [B,2,H,W]: ch0=avg, ch1=max
    *(f4*)(fb)       = a;
    *(f4*)(fb + HW_) = m;
}

// ----------------------------------------------------------------------------
// Kernel 2: 3x3 conv (2-in,1-out, pad 1) + sigmoid via one WMMA per 16 pixels.
// Implicit GEMM: D = A(16x32 f16, weights broadcast to all rows)
//                  x B(32x16 f16, im2col columns), K=18 real + 14 zero-pad.
// A layout (16-bit 16x32): lane L holds row M=L&15;
//   VGPR v, half p -> K = (v<4 ? 2v+p : 8+2v+p) + (L>=16 ? 8 : 0).
// B layout (16-bit 32x16): lane L holds column N=L&15;
//   half h -> K = (h<8 ? h : 8+h) + (L>=16 ? 8 : 0).
// D (f32 16x16): acc[0] on lanes 0-15 is row M=0 -> the single output channel.
// All loads are branch-free (clamped address + cndmask) so they issue as one
// batch with a single wait; pad slots K>=18 rely on A-weight == 0.
// ----------------------------------------------------------------------------
__global__ __launch_bounds__(256) void
sa_conv3x3_sigmoid_wmma(const float* __restrict__ feat,
                        const float* __restrict__ w,   // 18 floats [1,2,3,3]
                        float* __restrict__ out) {
    const int lane = threadIdx.x & 31;
    const int wave = threadIdx.x >> 5;
    const int tile = blockIdx.x * 8 + wave;          // 16 consecutive pixels
    const int n    = lane & 15;
    const int hi   = lane >> 4;                      // lane half-group
    const int p    = tile * 16 + n;                  // W=128 % 16 == 0: no row cross
    const int b    = p >> 14;
    const int ph   = (p >> 7) & 127;
    const int pw   = p & 127;

    // ---- A: conv weights, replicated across all 16 rows (pad K>=18 -> 0) ----
    v16h amat;
    #pragma unroll
    for (int h = 0; h < 16; ++h) {
        const int v  = h >> 1;
        const int pb = h & 1;
        const int K  = (v < 4 ? 2 * v + pb : 8 + 2 * v + pb) + hi * 8;
        const int Kc = (K < 18) ? K : 0;             // clamped, load unconditional
        const float wv = w[Kc];
        amat[h] = (K < 18) ? (_Float16)wv : (_Float16)0.0f;
    }

    // ---- B: im2col column for pixel p (branch-free clamped loads) ----
    const float* __restrict__ fb = feat + (size_t)b * 2 * HW_;
    v16h bmat;
    #pragma unroll
    for (int h = 0; h < 16; ++h) {
        const int K  = (h < 8 ? h : 8 + h) + hi * 8;
        const int Kc = (K < 18) ? K : 0;             // pad slots hit weight==0
        const int ci = (Kc >= 9) ? 1 : 0;
        const int r  = Kc - ci * 9;
        const int kh = r / 3;
        const int kw = r - kh * 3;
        const int y  = ph + kh - 1;
        const int x  = pw + kw - 1;
        const int yc = min(max(y, 0), H_ - 1);
        const int xc = min(max(x, 0), W_ - 1);
        const float v = fb[ci * HW_ + yc * W_ + xc]; // unconditional load
        const bool inb = (y == yc) & (x == xc);      // spatial zero-padding
        bmat[h] = inb ? (_Float16)v : (_Float16)0.0f;
    }

    // ---- D = A x B (EXEC is all-ones here: whole-wave launch) ----
    v8f acc = {};
    acc = __builtin_amdgcn_wmma_f32_16x16x32_f16(
        /*neg_a=*/false, amat, /*neg_b=*/false, bmat,
        /*c_mod=*/(short)0, acc, /*reuse_a=*/false, /*reuse_b=*/false);

    // Row M=0 lives in acc[0] on lanes 0-15.
    if (lane < 16) {
        const float yv = acc[0];
        // sigmoid via fast hardware rcp (v_exp_f32 + v_rcp_f32)
        out[p] = __builtin_amdgcn_rcpf(1.0f + __expf(-yv));
    }
}

// ----------------------------------------------------------------------------
extern "C" void kernel_launch(void* const* d_in, const int* in_sizes, int n_in,
                              void* d_out, int out_size, void* d_ws, size_t ws_size,
                              hipStream_t stream) {
    const float* x  = (const float*)d_in[0];   // [16,256,128,128] f32
    const float* cw = (const float*)d_in[1];   // [1,2,3,3] f32
    float* out  = (float*)d_out;               // [16,1,128,128] f32
    float* feat = (float*)d_ws;                // [16,2,128,128] f32 (2 MB scratch)

    // Stage 1: 65536 threads, 256/block -> 256 blocks.
    sa_reduce_mean_max<<<(NPIX / 4) / 256, 256, 0, stream>>>(x, feat);

    // Stage 2: 16384 tiles of 16 pixels, 8 waves (tiles) per 256-thread block.
    sa_conv3x3_sigmoid_wmma<<<(NPIX / 16) / 8, 256, 0, stream>>>(feat, cw, out);
    (void)in_sizes; (void)n_in; (void)out_size; (void)ws_size;
}
